// VQModel3D_55456617726516
// MI455X (gfx1250) — compile-verified, matrix operand order checked by reading, not run on hardware
//
#include <hip/hip_runtime.h>
#include <hip/hip_bf16.h>

// ---------------------------------------------------------------------------
// CDNA5 (gfx1250) VQ-VAE-3D forward pass.
// All 3x3x3 convs + deconvs + the 1x1x1 quant projection run as implicit GEMM on
// V_WMMA_F32_16X16X32_F16 (wave32).  Activations are channels-last f16 with
// channel stride padded to a multiple of 32 (8 for the packed first layer) so
// B-fragments are contiguous 16B loads per lane.  Weights are pre-swizzled
// into the A-fragment VGPR layout (ISA 7.12.2).  Input halos are staged in
// LDS and shared by all cout-tile waves of a workgroup.
// enc_in (CIN=8) packs the three x-taps into K (K = tx*8 + c) so each (tz,ty)
// tap-group is a single full-K WMMA instead of 3 quarter-utilized ones.
// ---------------------------------------------------------------------------

typedef __attribute__((ext_vector_type(16))) _Float16 v16h;
typedef __attribute__((ext_vector_type(8)))  float    v8f;

#define DEVINL __device__ __forceinline__

DEVINL float silu_f(float x) { return x / (1.0f + __expf(-x)); }

DEVINL v8f wmma32(v16h a, v16h b, v8f c) {
  // D = A(16x32 f16) * B(32x16 f16) + C(16x16 f32)
  return __builtin_amdgcn_wmma_f32_16x16x32_f16(
      /*neg_a=*/false, a, /*neg_b=*/false, b,
      /*c_mod=*/(short)0, c, /*reuse_a=*/false, /*reuse_b=*/false);
}

// B-fragment: lane L holds column n=L%16; k-half = L/16.
// b[0..7]  = In[n][khalf*8 + 0..7],  b[8..15] = In[n][16 + khalf*8 + 0..7]
DEVINL v16h load_bfrag(const _Float16* base, int khalf) {
  v16h bb;
  ((uint4*)&bb)[0] = *(const uint4*)(base + khalf * 8);
  ((uint4*)&bb)[1] = *(const uint4*)(base + 16 + khalf * 8);
  return bb;
}

// ---------------------------------------------------------------------------
// 1) bit-pack:  bool (B,1,256,256,256) -> mc f32 NCDHW (B,8,64^3)  (d_out)
//                                      -> mc f16 channels-last stride-8 (ws)
//    + per-16^3-patch occupancy mask (for zero_mask)
// ---------------------------------------------------------------------------
__global__ void k_pack(const unsigned char* __restrict__ x,
                       float* __restrict__ mc_out,
                       _Float16* __restrict__ mc8,
                       unsigned int* __restrict__ occ) {
  int tid = blockIdx.x * blockDim.x + threadIdx.x;
  if (tid >= 4 * 262144) return;
  int b = tid >> 18;
  int s = tid & 262143;
  int d = s >> 12, h = (s >> 6) & 63, w = s & 63;
  const unsigned char* xb = x + ((size_t)b << 24);
  unsigned int byteval[8];
#pragma unroll
  for (int i = 0; i < 8; ++i) byteval[i] = 0u;
#pragma unroll
  for (int dd = 0; dd < 4; ++dd) {
#pragma unroll
    for (int hh = 0; hh < 4; ++hh) {
      int iz = d * 4 + dd, iy = h * 4 + hh, ix = w * 4;
      const unsigned char* p = xb + ((size_t)iz << 16) + ((size_t)iy << 8) + ix;
      unsigned int nib = (p[0] ? 1u : 0u) | (p[1] ? 2u : 0u) |
                         (p[2] ? 4u : 0u) | (p[3] ? 8u : 0u);
      // q = dd*16 + hh*4 + ww ; channel = q>>3, bit = q&7
      byteval[dd * 2 + (hh >> 1)] |= nib << ((hh & 1) * 4);
    }
  }
  unsigned int any = 0;
#pragma unroll
  for (int i = 0; i < 8; ++i) any |= byteval[i];
  float* mo = mc_out + (((size_t)b * 8) << 18) + s;
  _Float16* m8 = mc8 + (size_t)tid * 8;
#pragma unroll
  for (int i = 0; i < 8; ++i) {
    float v = (float)byteval[i] * (1.0f / 255.0f);
    mo[(size_t)i << 18] = v;
    m8[i] = (_Float16)v;
  }
  if (any)
    atomicOr(&occ[(b << 12) | ((d >> 2) << 8) | ((h >> 2) << 4) | (w >> 2)], 1u);
}

// ---------------------------------------------------------------------------
// 2) weight repack: OIDHW f32 -> A-fragment f16 layout
//    frag[((t*MT+mo)*KC+kc)*32 + lane][e] = W[mo*16 + lane%16]
//                                            [kc*32 + (e/8)*16 + (lane/16)*8 + e%8][tap t]
// ---------------------------------------------------------------------------
__global__ void k_repack(const float* __restrict__ w, _Float16* __restrict__ frag,
                         int COUT, int CIN, int MT, int KC, int T) {
  int i = blockIdx.x * blockDim.x + threadIdx.x;
  int total = T * MT * KC * 512;
  if (i >= total) return;
  int e = i & 15;
  int lane = (i >> 4) & 31;
  int r = i >> 9;
  int kc = r % KC; r /= KC;
  int mo = r % MT;
  int t = r / MT;
  int m = mo * 16 + (lane & 15);
  int k = kc * 32 + ((e >> 3) << 4) + ((lane >> 4) << 3) + (e & 7);
  float v = (m < COUT && k < CIN) ? w[((size_t)m * CIN + k) * T + t] : 0.0f;
  frag[i] = (_Float16)v;
}

// repack for enc_in tap-packed layout: K = tx*8 + c (tx<3), tap group t9=(tz,ty)
// frag[((t9*2 + mo)*32 + lane)*16 + e]
__global__ void k_repack_in(const float* __restrict__ w, _Float16* __restrict__ frag) {
  int i = blockIdx.x * blockDim.x + threadIdx.x;
  if (i >= 9 * 2 * 512) return;
  int e = i & 15;
  int lane = (i >> 4) & 31;
  int r = i >> 9;
  int mo = r & 1;
  int t9 = r >> 1;
  int m = mo * 16 + (lane & 15);
  int k = ((e >> 3) << 4) + ((lane >> 4) << 3) + (e & 7);
  int tx = k >> 3, c = k & 7;
  // w: (32, 8, 3,3,3) OIDHW ; tap = t9*3 + tx
  float v = (tx < 3) ? w[((size_t)m * 8 + c) * 27 + t9 * 3 + tx] : 0.0f;
  frag[i] = (_Float16)v;
}

// ---------------------------------------------------------------------------
// 3) enc_in conv: CIN=8 (stride-8 input), COUT=32, stride 1, silu.
//    Three x-taps packed into K -> 9 WMMAs per 16x16 output tile.
// ---------------------------------------------------------------------------
__global__ void __launch_bounds__(64)
k_conv3_in(const _Float16* __restrict__ in, const _Float16* __restrict__ frag,
           const float* __restrict__ bias, _Float16* __restrict__ out, int D) {
  constexpr int IX = 18;
  __shared__ _Float16 lds[9 * IX * 8];
  const int tid = threadIdx.x;
  const int wv = tid >> 5;
  const int lane = tid & 31;
  const int XT = D >> 4;
  int bid = blockIdx.x;
  const int xt = bid % XT; bid /= XT;
  const int oy = bid % D;
  const int oz = bid / D;
  const int b = blockIdx.y;
  const int x0 = xt << 4;

  for (int c = tid; c < 9 * IX; c += 64) {
    const int lx = c % IX;
    const int sy = (c / IX) % 3;
    const int sz = c / (IX * 3);
    const int ix = x0 + lx - 1;
    const int iy = oy - 1 + sy;
    const int iz = oz - 1 + sz;
    uint4 val = {0u, 0u, 0u, 0u};
    if (ix >= 0 && ix < D && iy >= 0 && iy < D && iz >= 0 && iz < D) {
      val = *(const uint4*)(in + ((((size_t)b * D + iz) * D + iy) * D + ix) * 8);
    }
    *(uint4*)(&lds[(size_t)c * 8]) = val;
  }
  __syncthreads();

  v8f acc = {};
  const int n = lane & 15;
  const int khalf = lane >> 4;
#pragma unroll
  for (int t9 = 0; t9 < 9; ++t9) {
    v16h a = *(const v16h*)(frag + (((size_t)t9 * 2 + wv) * 32 + lane) * 16);
    v16h bb;
    // low half:  k-group = khalf      -> tap tx = khalf, 8 channels
    // high half: k-group = 2 + khalf  -> tap tx = 2 (khalf==0) else zero-pad
    ((uint4*)&bb)[0] = *(const uint4*)(&lds[((size_t)t9 * IX + (n + khalf)) * 8]);
    uint4 hi = *(const uint4*)(&lds[((size_t)t9 * IX + (n + 2)) * 8]);
    if (khalf != 0) { hi.x = 0u; hi.y = 0u; hi.z = 0u; hi.w = 0u; }
    ((uint4*)&bb)[1] = hi;
    acc = wmma32(a, bb, acc);
  }

  const int ox = x0 + n;
  _Float16* op = out +
      ((((size_t)b * D + oz) * D + oy) * D + ox) * 32 + (wv << 4);
#pragma unroll
  for (int v = 0; v < 8; ++v) {
    const int m = (khalf << 3) + v;
    float val = acc[v] + bias[(wv << 4) + m];
    op[m] = (_Float16)silu_f(val);
  }
}

// ---------------------------------------------------------------------------
// 4) generic 3x3x3 conv (stride 1 or 2), channels-last f16 in/out, WMMA.
// ---------------------------------------------------------------------------
template <int CPAD, int COUT, int STRIDE, bool SILU, int WAVES>
__global__ void __launch_bounds__(WAVES * 32)
k_conv3(const _Float16* __restrict__ in, const _Float16* __restrict__ frag,
        const float* __restrict__ bias, _Float16* __restrict__ out,
        int Din, int Dout) {
  constexpr int IX = 16 * STRIDE + 2;
  constexpr int KC = CPAD / 32;
  constexpr int MT = WAVES;
  constexpr int CH16 = CPAD / 8;  // 16B chunks per site
  __shared__ _Float16 lds[3 * 3 * IX * CPAD];

  const int tid = threadIdx.x;
  const int wv = tid >> 5;
  const int lane = tid & 31;
  const int XT = Dout >> 4;
  int bid = blockIdx.x;
  const int xt = bid % XT; bid /= XT;
  const int oy = bid % Dout;
  const int oz = bid / Dout;
  const int b = blockIdx.y;
  const int x0 = xt << 4;

  const int nchunks = 9 * IX * CH16;
  for (int c = tid; c < nchunks; c += WAVES * 32) {
    const int cc = c % CH16;
    const int site = c / CH16;
    const int lx = site % IX;
    const int sy = (site / IX) % 3;
    const int sz = site / (IX * 3);
    const int ix = x0 * STRIDE + lx - 1;
    const int iy = oy - 1 + sy;
    const int iz = oz - 1 + sz;
    uint4 val = {0u, 0u, 0u, 0u};
    if (ix >= 0 && ix < Din && iy >= 0 && iy < Din && iz >= 0 && iz < Din) {
      val = *(const uint4*)(in +
            ((((size_t)b * Din + iz) * Din + iy) * Din + ix) * CPAD + cc * 8);
    }
    *(uint4*)(&lds[(size_t)site * CPAD + cc * 8]) = val;
  }
  __syncthreads();

  v8f acc = {};
  const int n = lane & 15;
  const int khalf = lane >> 4;
  for (int t = 0; t < 27; ++t) {
    const int tx = t % 3;
    const int ty = (t / 3) % 3;
    const int tz = t / 9;
    const _Float16* bbase =
        &lds[((size_t)(tz * 3 + ty) * IX + (n * STRIDE + tx)) * CPAD];
#pragma unroll
    for (int kc = 0; kc < KC; ++kc) {
      v16h a = *(const v16h*)(frag +
                ((((size_t)t * MT + wv) * KC + kc) * 32 + lane) * 16);
      v16h bb = load_bfrag(bbase + kc * 32, khalf);
      acc = wmma32(a, bb, acc);
    }
  }

  const int ox = x0 + n;
  _Float16* op = out +
      ((((size_t)b * Dout + oz) * Dout + oy) * Dout + ox) * COUT + (wv << 4);
#pragma unroll
  for (int v = 0; v < 8; ++v) {
    const int m = (khalf << 3) + v;
    float val = acc[v] + bias[(wv << 4) + m];
    if (SILU) val = silu_f(val);
    op[m] = (_Float16)val;
  }
}

// ---------------------------------------------------------------------------
// 5) fractionally-strided conv (lhs_dilation=2, padding (2,1)) via 8 output
//    parity classes.  Parity p per dim: p==0 -> taps {w0@in[i-1], w2@in[i]},
//    p==1 -> {w1@in[i]}.  Each block covers 16 same-parity x outputs.
// ---------------------------------------------------------------------------
template <int CPAD, int COUT, bool SILU, int WAVES>
__global__ void __launch_bounds__(WAVES * 32)
k_deconv(const _Float16* __restrict__ in, const _Float16* __restrict__ frag,
         const float* __restrict__ bias, _Float16* __restrict__ out, int Din) {
  constexpr int IX = 17;
  constexpr int KC = CPAD / 32;
  constexpr int MT = WAVES;
  constexpr int CH16 = CPAD / 8;
  __shared__ _Float16 lds[2 * 2 * IX * CPAD];
  const int Dout = Din * 2;
  const int tid = threadIdx.x;
  const int wv = tid >> 5;
  const int lane = tid & 31;
  const int XT = Dout >> 4;
  int bid = blockIdx.x;
  const int xtp = bid % XT; bid /= XT;
  const int oy = bid % Dout;
  const int oz = bid / Dout;
  const int b = blockIdx.y;
  const int px = xtp & 1, xt = xtp >> 1;
  const int py = oy & 1, iy0 = oy >> 1;
  const int pz = oz & 1, iz0 = oz >> 1;
  const int x0 = xt << 4;

  const int nchunks = 4 * IX * CH16;
  for (int c = tid; c < nchunks; c += WAVES * 32) {
    const int cc = c % CH16;
    const int site = c / CH16;
    const int lx = site % IX;
    const int sy = (site / IX) & 1;
    const int sz = site / (IX * 2);
    const int ix = x0 + lx - 1;
    const int iy = iy0 - 1 + sy;
    const int iz = iz0 - 1 + sz;
    uint4 val = {0u, 0u, 0u, 0u};
    if (ix >= 0 && ix < Din && iy >= 0 && iy < Din && iz >= 0 && iz < Din) {
      val = *(const uint4*)(in +
            ((((size_t)b * Din + iz) * Din + iy) * Din + ix) * CPAD + cc * 8);
    }
    *(uint4*)(&lds[(size_t)site * CPAD + cc * 8]) = val;
  }
  __syncthreads();

  v8f acc = {};
  const int n = lane & 15;
  const int khalf = lane >> 4;
  const int cz = pz ? 1 : 2, cy = py ? 1 : 2, cx = px ? 1 : 2;
  for (int jz = 0; jz < cz; ++jz) {
    const int sz = pz ? 1 : jz;
    const int wtz = pz ? 1 : jz * 2;
    for (int jy = 0; jy < cy; ++jy) {
      const int sy = py ? 1 : jy;
      const int wty = py ? 1 : jy * 2;
      for (int jx = 0; jx < cx; ++jx) {
        const int sx = px ? 1 : jx;
        const int wtx = px ? 1 : jx * 2;
        const int t = (wtz * 3 + wty) * 3 + wtx;
        const _Float16* bbase =
            &lds[((size_t)(sz * 2 + sy) * IX + (n + sx)) * CPAD];
#pragma unroll
        for (int kc = 0; kc < KC; ++kc) {
          v16h a = *(const v16h*)(frag +
                    ((((size_t)t * MT + wv) * KC + kc) * 32 + lane) * 16);
          v16h bb = load_bfrag(bbase + kc * 32, khalf);
          acc = wmma32(a, bb, acc);
        }
      }
    }
  }
  const int ox = 2 * (x0 + n) + px;
  _Float16* op = out +
      ((((size_t)b * Dout + oz) * Dout + oy) * Dout + ox) * COUT + (wv << 4);
#pragma unroll
  for (int v = 0; v < 8; ++v) {
    const int m = (khalf << 3) + v;
    float val = acc[v] + bias[(wv << 4) + m];
    if (SILU) val = silu_f(val);
    op[m] = (_Float16)val;
  }
}

// ---------------------------------------------------------------------------
// 6) final conv: 32 -> 8 channels, writes f32 NCDHW directly to d_out.
// ---------------------------------------------------------------------------
__global__ void __launch_bounds__(32)
k_conv3_out(const _Float16* __restrict__ in, const _Float16* __restrict__ frag,
            const float* __restrict__ bias, float* __restrict__ recon, int D) {
  constexpr int IX = 18;
  constexpr int CPAD = 32;
  constexpr int CH16 = CPAD / 8;
  __shared__ _Float16 lds[3 * 3 * IX * CPAD];
  const int tid = threadIdx.x;
  const int lane = tid & 31;
  const int XT = D >> 4;
  int bid = blockIdx.x;
  const int xt = bid % XT; bid /= XT;
  const int oy = bid % D;
  const int oz = bid / D;
  const int b = blockIdx.y;
  const int x0 = xt << 4;

  const int nchunks = 9 * IX * CH16;
  for (int c = tid; c < nchunks; c += 32) {
    const int cc = c % CH16;
    const int site = c / CH16;
    const int lx = site % IX;
    const int sy = (site / IX) % 3;
    const int sz = site / (IX * 3);
    const int ix = x0 + lx - 1;
    const int iy = oy - 1 + sy;
    const int iz = oz - 1 + sz;
    uint4 val = {0u, 0u, 0u, 0u};
    if (ix >= 0 && ix < D && iy >= 0 && iy < D && iz >= 0 && iz < D) {
      val = *(const uint4*)(in +
            ((((size_t)b * D + iz) * D + iy) * D + ix) * CPAD + cc * 8);
    }
    *(uint4*)(&lds[(size_t)site * CPAD + cc * 8]) = val;
  }
  __syncthreads();

  v8f acc = {};
  const int n = lane & 15;
  const int khalf = lane >> 4;
  for (int t = 0; t < 27; ++t) {
    const int tx = t % 3;
    const int ty = (t / 3) % 3;
    const int tz = t / 9;
    const _Float16* bbase = &lds[((size_t)(tz * 3 + ty) * IX + (n + tx)) * CPAD];
    v16h a = *(const v16h*)(frag + ((size_t)t * 32 + lane) * 16);
    v16h bb = load_bfrag(bbase, khalf);
    acc = wmma32(a, bb, acc);
  }
  const int ox = x0 + n;
  const size_t sp = ((size_t)oz * D + oy) * D + ox;
  const size_t vol = (size_t)D * D * D;
#pragma unroll
  for (int v = 0; v < 8; ++v) {
    const int m = (khalf << 3) + v;
    if (m < 8) recon[((size_t)b * 8 + m) * vol + sp] = acc[v] + bias[m];
  }
}

// ---------------------------------------------------------------------------
// 7) quant 1x1x1 conv (128 -> E=16) as WMMA GEMM; writes f32 seq[token][16]
// ---------------------------------------------------------------------------
__global__ void __launch_bounds__(32)
k_quant(const _Float16* __restrict__ h4, const _Float16* __restrict__ frag,
        const float* __restrict__ qb, float* __restrict__ seq) {
  const int lane = threadIdx.x & 31;
  const int tok0 = blockIdx.x * 16;
  const int n = lane & 15;
  const int khalf = lane >> 4;
  v8f acc = {};
  const _Float16* bbase = h4 + (size_t)(tok0 + n) * 128;
#pragma unroll
  for (int kc = 0; kc < 4; ++kc) {
    v16h a = *(const v16h*)(frag + ((size_t)kc * 32 + lane) * 16);
    v16h bb = load_bfrag(bbase + kc * 32, khalf);
    acc = wmma32(a, bb, acc);
  }
  float* op = seq + (size_t)(tok0 + n) * 16;
#pragma unroll
  for (int v = 0; v < 8; ++v) {
    const int m = (khalf << 3) + v;
    op[m] = acc[v] + qb[m];
  }
}

// ---------------------------------------------------------------------------
// 8) VQ: per-token argmin over 8192 codes (|c|^2 - 2 x.c), wave32 reduction.
// ---------------------------------------------------------------------------
__global__ void k_vq(const float* __restrict__ seq, const float* __restrict__ cb,
                     const unsigned int* __restrict__ occ, int* __restrict__ idxbuf,
                     float* __restrict__ fidx_out, float* __restrict__ commit_acc) {
  const int gwave = (blockIdx.x * blockDim.x + threadIdx.x) >> 5;
  const int lane = threadIdx.x & 31;
  if (gwave >= 16384) return;
  const float* xv = seq + (size_t)gwave * 16;
  float x[16];
#pragma unroll
  for (int e = 0; e < 16; ++e) x[e] = xv[e];
  float best = 1e30f;
  int bidx = 0;
  for (int k = lane; k < 8192; k += 32) {
    const float* c = cb + (size_t)k * 16;
    float d = 0.0f;
#pragma unroll
    for (int e = 0; e < 16; ++e) {
      float ce = c[e];
      d += ce * (ce - 2.0f * x[e]);
    }
    if (d < best) { best = d; bidx = k; }
  }
#pragma unroll
  for (int off = 16; off > 0; off >>= 1) {
    float ob = __shfl_xor(best, off, 32);
    int oi = __shfl_xor(bidx, off, 32);
    if (ob < best || (ob == best && oi < bidx)) { best = ob; bidx = oi; }
  }
  if (lane == 0) {
    const bool zero = (occ[gwave] == 0u);
    const int fi = zero ? 0 : bidx + 1;
    idxbuf[gwave] = fi;
    fidx_out[gwave] = (float)fi;
    if (!zero) {
      const float* c = cb + (size_t)bidx * 16;
      float mse = 0.0f;
#pragma unroll
      for (int e = 0; e < 16; ++e) {
        float df = c[e] - x[e];
        mse += df * df;
      }
      atomicAdd(commit_acc, mse * (1.0f / 16.0f));
    }
  }
}

__global__ void k_commit(const float* __restrict__ acc, float* __restrict__ out) {
  out[0] = 0.25f * acc[0] * (1.0f / 16384.0f);
}

// ---------------------------------------------------------------------------
// 9) build decoder input: quant = full_cb[full_idx] + pos_emb, then 1x1x1
//    post-quant projection (16 -> 128), channels-last f16.
// ---------------------------------------------------------------------------
__global__ void k_buildq(const int* __restrict__ idxbuf, const float* __restrict__ cb,
                         const float* __restrict__ repl, const float* __restrict__ pe,
                         const float* __restrict__ pqw, const float* __restrict__ pqb,
                         _Float16* __restrict__ qin) {
  const int gwave = (blockIdx.x * blockDim.x + threadIdx.x) >> 5;
  const int lane = threadIdx.x & 31;
  if (gwave >= 16384) return;
  const int sp = gwave & 4095;
  const int fi = idxbuf[gwave];
  const float* src = (fi == 0) ? repl : cb + (size_t)(fi - 1) * 16;
  float v[16];
#pragma unroll
  for (int e = 0; e < 16; ++e) v[e] = src[e] + pe[(size_t)e * 4096 + sp];
  _Float16* o = qin + (size_t)gwave * 128;
  for (int c = lane; c < 128; c += 32) {
    float s = pqb[c];
    const float* wr = pqw + (size_t)c * 16;
#pragma unroll
    for (int e = 0; e < 16; ++e) s += wr[e] * v[e];
    o[c] = (_Float16)s;
  }
}

// ---------------------------------------------------------------------------
// host-side launch
// ---------------------------------------------------------------------------
static inline void launch_repack(const float* w, _Float16* frag, int COUT,
                                 int CIN, int MT, int KC, int T, hipStream_t s) {
  int total = T * MT * KC * 512;
  k_repack<<<(total + 255) / 256, 256, 0, s>>>(w, frag, COUT, CIN, MT, KC, T);
}

extern "C" void kernel_launch(void* const* d_in, const int* in_sizes, int n_in,
                              void* d_out, int out_size, void* d_ws, size_t ws_size,
                              hipStream_t stream) {
  (void)in_sizes; (void)n_in; (void)out_size; (void)ws_size;
  const unsigned char* x = (const unsigned char*)d_in[0];
  const float* enc_in_w  = (const float*)d_in[1];
  const float* enc_in_b  = (const float*)d_in[2];
  const float* enc_d1_w  = (const float*)d_in[3];
  const float* enc_d1_b  = (const float*)d_in[4];
  const float* enc_d2_w  = (const float*)d_in[5];
  const float* enc_d2_b  = (const float*)d_in[6];
  const float* enc_out_w = (const float*)d_in[7];
  const float* enc_out_b = (const float*)d_in[8];
  const float* quant_w   = (const float*)d_in[9];
  const float* quant_b   = (const float*)d_in[10];
  const float* codebook  = (const float*)d_in[11];
  const float* repl      = (const float*)d_in[12];
  const float* pos_emb   = (const float*)d_in[13];
  const float* pq_w      = (const float*)d_in[14];
  const float* pq_b      = (const float*)d_in[15];
  const float* dec_in_w  = (const float*)d_in[16];
  const float* dec_in_b  = (const float*)d_in[17];
  const float* dec_u1_w  = (const float*)d_in[18];
  const float* dec_u1_b  = (const float*)d_in[19];
  const float* dec_u2_w  = (const float*)d_in[20];
  const float* dec_u2_b  = (const float*)d_in[21];
  const float* dec_out_w = (const float*)d_in[22];
  const float* dec_out_b = (const float*)d_in[23];

  float* out = (float*)d_out;
  float* recon      = out;                 // (4,8,64,64,64)
  float* mc_out     = out + 8388608;       // (4,8,64,64,64)
  float* commit_out = out + 16777216;      // scalar
  float* fidx_out   = out + 16777217;      // (4,4096)

  char* ws = (char*)d_ws;
  size_t off = 0;
  auto alloc = [&](size_t bytes) -> void* {
    off = (off + 255) & ~(size_t)255;
    void* p = ws + off;
    off += bytes;
    return p;
  };
  unsigned int* occ = (unsigned int*)alloc(16384 * 4);
  float* commit_acc = (float*)alloc(256);
  int* idxbuf       = (int*)alloc(16384 * 4);
  float* seq        = (float*)alloc((size_t)16384 * 16 * 4);
  _Float16* mc8  = (_Float16*)alloc((size_t)4 * 262144 * 8 * 2);   // stride-8
  _Float16* h1   = (_Float16*)alloc((size_t)4 * 262144 * 32 * 2);
  _Float16* h2   = (_Float16*)alloc((size_t)4 * 32768 * 64 * 2);
  _Float16* h3   = (_Float16*)alloc((size_t)4 * 4096 * 128 * 2);
  _Float16* h4   = (_Float16*)alloc((size_t)4 * 4096 * 128 * 2);
  _Float16* qin  = (_Float16*)alloc((size_t)4 * 4096 * 128 * 2);
  _Float16* u2   = (_Float16*)alloc((size_t)4 * 262144 * 32 * 2);
  _Float16* dd1 = h3;   // alias: h3 dead after enc_out
  _Float16* u1  = h2;   // alias: h2 dead after enc_d2
  auto fragalloc = [&](int T, int MT, int KC) {
    return (_Float16*)alloc((size_t)T * MT * KC * 512 * 2);
  };
  _Float16* f_enc_in  = (_Float16*)alloc((size_t)9 * 2 * 512 * 2);  // tap-packed
  _Float16* f_enc_d1  = fragalloc(27, 4, 1);
  _Float16* f_enc_d2  = fragalloc(27, 8, 2);
  _Float16* f_enc_out = fragalloc(27, 8, 4);
  _Float16* f_quant   = fragalloc(1, 1, 4);
  _Float16* f_dec_in  = fragalloc(27, 8, 4);
  _Float16* f_dec_u1  = fragalloc(27, 4, 4);
  _Float16* f_dec_u2  = fragalloc(27, 2, 2);
  _Float16* f_dec_out = fragalloc(27, 1, 1);

  hipMemsetAsync(occ, 0, 16384 * 4, stream);
  hipMemsetAsync(commit_acc, 0, 4, stream);

  k_repack_in<<<(9 * 2 * 512 + 255) / 256, 256, 0, stream>>>(enc_in_w, f_enc_in);
  launch_repack(enc_d1_w,  f_enc_d1,  64,  32,  4, 1, 27, stream);
  launch_repack(enc_d2_w,  f_enc_d2,  128, 64,  8, 2, 27, stream);
  launch_repack(enc_out_w, f_enc_out, 128, 128, 8, 4, 27, stream);
  launch_repack(quant_w,   f_quant,   16,  128, 1, 4, 1,  stream);
  launch_repack(dec_in_w,  f_dec_in,  128, 128, 8, 4, 27, stream);
  launch_repack(dec_u1_w,  f_dec_u1,  64,  128, 4, 4, 27, stream);
  launch_repack(dec_u2_w,  f_dec_u2,  32,  64,  2, 2, 27, stream);
  launch_repack(dec_out_w, f_dec_out, 8,   32,  1, 1, 27, stream);

  k_pack<<<4096, 256, 0, stream>>>(x, mc_out, mc8, occ);

  // encoder
  { dim3 g(4 * 64 * 64, 4);
    k_conv3_in<<<g, 64, 0, stream>>>(mc8, f_enc_in, enc_in_b, h1, 64); }
  { dim3 g(2 * 32 * 32, 4);
    k_conv3<32, 64, 2, true, 4><<<g, 128, 0, stream>>>(h1, f_enc_d1, enc_d1_b, h2, 64, 32); }
  { dim3 g(1 * 16 * 16, 4);
    k_conv3<64, 128, 2, true, 8><<<g, 256, 0, stream>>>(h2, f_enc_d2, enc_d2_b, h3, 32, 16); }
  { dim3 g(1 * 16 * 16, 4);
    k_conv3<128, 128, 1, false, 8><<<g, 256, 0, stream>>>(h3, f_enc_out, enc_out_b, h4, 16, 16); }

  // quantize
  k_quant<<<1024, 32, 0, stream>>>(h4, f_quant, quant_b, seq);
  k_vq<<<2048, 256, 0, stream>>>(seq, codebook, occ, idxbuf, fidx_out, commit_acc);
  k_commit<<<1, 1, 0, stream>>>(commit_acc, commit_out);
  k_buildq<<<2048, 256, 0, stream>>>(idxbuf, codebook, repl, pos_emb, pq_w, pq_b, qin);

  // decoder
  { dim3 g(1 * 16 * 16, 4);
    k_conv3<128, 128, 1, true, 8><<<g, 256, 0, stream>>>(qin, f_dec_in, dec_in_b, dd1, 16, 16); }
  { dim3 g(2 * 32 * 32, 4);
    k_deconv<128, 64, true, 4><<<g, 128, 0, stream>>>(dd1, f_dec_u1, dec_u1_b, u1, 16); }
  { dim3 g(4 * 64 * 64, 4);
    k_deconv<64, 32, true, 2><<<g, 64, 0, stream>>>(u1, f_dec_u2, dec_u2_b, u2, 32); }
  { dim3 g(4 * 64 * 64, 4);
    k_conv3_out<<<g, 32, 0, stream>>>(u2, f_dec_out, dec_out_b, recon, 64); }
}
